// SetConvDecoder_30872224924408
// MI455X (gfx1250) — matrix operand order repaired and checked by reading, not compiled, not run on hardware
//
#include <hip/hip_runtime.h>
#include <hip/hip_bf16.h>

typedef __attribute__((ext_vector_type(16))) _Float16 v16h;
typedef __attribute__((ext_vector_type(4)))  _Float16 v4h;
typedef __attribute__((ext_vector_type(8)))  float    v8f;

#define BATCH   4
#define NGRID   16384            // 128*128
#define NTGT    2048
#define DZV     128
#define KC      32               // K per WMMA chunk
#define NCHUNK  (NGRID / KC)     // 512
#define MT      32               // target rows per workgroup (two 16-row M tiles)
#define AS_STRIDE 40             // halves: 80B row -> 16B-aligned b128 fragment loads
#define ZH_BYTES ((size_t)BATCH * NGRID * DZV * 2)   // 16 MB packed f16 z

// ---------------------------------------------------------------------------
// Kernel 1: one-time pre-pack of z_grid (f32, g-major) into fragment-ready f16.
// Layout: zh[b][chunk][ntile][lane][16 halves]; lane holds column (lane&15) of
// the 32x16 B tile, halves h -> K = 8*(lane>=16) + h + 8*(h>=8)  (CDNA5 16-bit B layout).
// ---------------------------------------------------------------------------
__global__ __launch_bounds__(256)
void zpack_kernel(const float* __restrict__ zg, _Float16* __restrict__ zh)
{
    __shared__ float zt[KC][DZV + 4];     // 32 x 132 f32, padded

    const int b = blockIdx.x >> 9;        // / NCHUNK
    const int c = blockIdx.x & (NCHUNK - 1);
    const int tid = threadIdx.x;

    // load phase: 32 rows x 128 f32, coalesced (each thread: 16 contiguous f32)
    {
        const int row = tid >> 3;
        const int seg = (tid & 7) * 16;
        const float* zp = zg + ((size_t)b * NGRID + (size_t)c * KC + row) * DZV + seg;
        #pragma unroll
        for (int i = 0; i < 16; ++i) zt[row][seg + i] = zp[i];
    }
    __syncthreads();

    // pack phase: wave w -> N-tile w; lane gathers its fragment column
    {
        const int w    = tid >> 5;
        const int lane = tid & 31;
        const int col  = lane & 15;
        const int kb   = (lane >> 4) * 8;
        v16h frag;
        #pragma unroll
        for (int h = 0; h < 16; ++h)
            frag[h] = (_Float16)zt[kb + h + ((h >= 8) ? 8 : 0)][w * 16 + col];
        _Float16* op = zh + ((((size_t)b * NCHUNK + c) * 8 + w) * 32 + lane) * 16;
        *(v16h*)op = frag;
    }
}

// ---------------------------------------------------------------------------
// Kernel 2: main WMMA kernel. B fragments come straight from packed zh global;
// A (exp weights) double-buffered in LDS with one barrier per K-chunk.
// ---------------------------------------------------------------------------
__global__ __launch_bounds__(256)
void setconv_wmma_kernel(const float* __restrict__ xg,     // (B, NG, 2)
                         const _Float16* __restrict__ zh,  // packed f16 z
                         const float* __restrict__ xt,     // (B, NT, 2)
                         const float* __restrict__ lsp,    // (2,)
                         float* __restrict__ out)          // (B, NT, 128)
{
    __shared__ __align__(16) _Float16 As[2][MT * AS_STRIDE];   // weights M x K

    const int tid    = threadIdx.x;
    const int wave   = tid >> 5;          // 0..7 -> N tile
    const int lane   = tid & 31;
    const int laneHi = (lane >> 4) & 1;
    const int col    = lane & 15;

    const int b  = blockIdx.x >> 6;       // 64 row-tiles per batch
    const int t0 = (blockIdx.x & 63) * MT;

    const float inv0 = 1.0f / (1e-5f + log1pf(__expf(lsp[0])));
    const float inv1 = 1.0f / (1e-5f + log1pf(__expf(lsp[1])));

    // weight staging: thread -> (row wm, 4 consecutive k)
    const int wm = tid >> 3;              // 0..31
    const int wk = (tid & 7) * 4;         // 0,4,...,28
    const float tx0 = xt[((size_t)b * NTGT + t0 + wm) * 2 + 0] * inv0;
    const float tx1 = xt[((size_t)b * NTGT + t0 + wm) * 2 + 1] * inv1;

    const float*    __restrict__ xg_b = xg + (size_t)b * NGRID * 2;
    const _Float16* __restrict__ zh_b = zh + ((size_t)b * NCHUNK * 8) * 512;

    // stage chunk 0 weights into As[0]
    {
        const float4* xp = (const float4*)(xg_b + (size_t)wk * 2);
        float4 p0 = xp[0], p1 = xp[1];
        v4h w;
        float d0, d1;
        d0 = tx0 - p0.x * inv0; d1 = tx1 - p0.y * inv1;
        w[0] = (_Float16)__expf(-0.5f * (d0 * d0 + d1 * d1));
        d0 = tx0 - p0.z * inv0; d1 = tx1 - p0.w * inv1;
        w[1] = (_Float16)__expf(-0.5f * (d0 * d0 + d1 * d1));
        d0 = tx0 - p1.x * inv0; d1 = tx1 - p1.y * inv1;
        w[2] = (_Float16)__expf(-0.5f * (d0 * d0 + d1 * d1));
        d0 = tx0 - p1.z * inv0; d1 = tx1 - p1.w * inv1;
        w[3] = (_Float16)__expf(-0.5f * (d0 * d0 + d1 * d1));
        *(v4h*)&As[0][wm * AS_STRIDE + wk] = w;
    }
    __syncthreads();

    v8f acc0 = {};   // rows t0 .. t0+15
    v8f acc1 = {};   // rows t0+16 .. t0+31

    const int abase = col * AS_STRIDE + laneHi * 8;   // fragment K-base (halves)

    for (int c = 0; c < NCHUNK; ++c) {
        const int p = c & 1;

        // B fragment: one 32B coalesced read per lane from packed zh
        v16h bf = *(const v16h*)(zh_b + (((size_t)c * 8 + wave) * 32 + lane) * 16);

        // A fragments: two aligned 16B LDS reads each (K 0-7/16-23 + laneHi*8)
        v16h a0, a1;
        {
            uint4* a0q = reinterpret_cast<uint4*>(&a0);
            uint4* a1q = reinterpret_cast<uint4*>(&a1);
            a0q[0] = *(const uint4*)&As[p][abase];
            a0q[1] = *(const uint4*)&As[p][abase + 16];
            a1q[0] = *(const uint4*)&As[p][abase + 16 * AS_STRIDE];
            a1q[1] = *(const uint4*)&As[p][abase + 16 * AS_STRIDE + 16];
        }

        // stage next chunk's weights into the other buffer (overlaps WMMA)
        if (c + 1 < NCHUNK) {
            const float4* xp = (const float4*)(xg_b + ((size_t)(c + 1) * KC + wk) * 2);
            float4 p0 = xp[0], p1 = xp[1];
            v4h w;
            float d0, d1;
            d0 = tx0 - p0.x * inv0; d1 = tx1 - p0.y * inv1;
            w[0] = (_Float16)__expf(-0.5f * (d0 * d0 + d1 * d1));
            d0 = tx0 - p0.z * inv0; d1 = tx1 - p0.w * inv1;
            w[1] = (_Float16)__expf(-0.5f * (d0 * d0 + d1 * d1));
            d0 = tx0 - p1.x * inv0; d1 = tx1 - p1.y * inv1;
            w[2] = (_Float16)__expf(-0.5f * (d0 * d0 + d1 * d1));
            d0 = tx0 - p1.z * inv0; d1 = tx1 - p1.w * inv1;
            w[3] = (_Float16)__expf(-0.5f * (d0 * d0 + d1 * d1));
            *(v4h*)&As[p ^ 1][wm * AS_STRIDE + wk] = w;
        }

        acc0 = __builtin_amdgcn_wmma_f32_16x16x32_f16(false, a0, false, bf,
                                                      (short)0, acc0, false, false);
        acc1 = __builtin_amdgcn_wmma_f32_16x16x32_f16(false, a1, false, bf,
                                                      (short)0, acc1, false, false);
        __syncthreads();
    }

    // write C tiles: VGPR r -> M = r + 8*laneHi ; N = col ; scale 1/128
    const float scale = 1.0f / 128.0f;
    float* op = out + ((size_t)b * NTGT + t0) * DZV + wave * 16 + col;
    #pragma unroll
    for (int r = 0; r < 8; ++r) {
        const int m = r + laneHi * 8;
        op[(size_t)m        * DZV] = acc0[r] * scale;
        op[(size_t)(m + 16) * DZV] = acc1[r] * scale;
    }
}

// ---------------------------------------------------------------------------
// Fallback (round-1 kernel): used only if workspace is too small to pre-pack.
// ---------------------------------------------------------------------------
#define BS_STRIDE 36
__global__ __launch_bounds__(256)
void setconv_wmma_fallback(const float* __restrict__ xg, const float* __restrict__ zg,
                           const float* __restrict__ xt, const float* __restrict__ lsp,
                           float* __restrict__ out)
{
    __shared__ __align__(16) _Float16 Aw[MT * AS_STRIDE];
    __shared__ __align__(16) _Float16 Bs[DZV * BS_STRIDE];

    const int tid = threadIdx.x, wave = tid >> 5, lane = tid & 31;
    const int laneHi = (lane >> 4) & 1, col = lane & 15;
    const int b = blockIdx.x >> 6, t0 = (blockIdx.x & 63) * MT;

    const float inv0 = 1.0f / (1e-5f + log1pf(__expf(lsp[0])));
    const float inv1 = 1.0f / (1e-5f + log1pf(__expf(lsp[1])));
    const int wm = tid >> 3, wk = (tid & 7) * 4;
    const float tx0 = xt[((size_t)b * NTGT + t0 + wm) * 2 + 0] * inv0;
    const float tx1 = xt[((size_t)b * NTGT + t0 + wm) * 2 + 1] * inv1;
    const int zk = tid >> 3, zn = (tid & 7) * 16;
    const float* xg_b = xg + (size_t)b * NGRID * 2;
    const float* zg_b = zg + (size_t)b * NGRID * DZV;

    v8f acc0 = {}, acc1 = {};
    for (int g0 = 0; g0 < NGRID; g0 += KC) {
        const float* xp = xg_b + (size_t)(g0 + wk) * 2;
        #pragma unroll
        for (int i = 0; i < 4; ++i) {
            float d0 = tx0 - xp[2 * i + 0] * inv0;
            float d1 = tx1 - xp[2 * i + 1] * inv1;
            Aw[wm * AS_STRIDE + wk + i] = (_Float16)__expf(-0.5f * (d0 * d0 + d1 * d1));
        }
        const float* zp = zg_b + (size_t)(g0 + zk) * DZV + zn;
        #pragma unroll
        for (int i = 0; i < 16; ++i)
            Bs[(zn + i) * BS_STRIDE + zk] = (_Float16)zp[i];
        __syncthreads();

        v16h a0, a1, bf;
        unsigned int* a0u = reinterpret_cast<unsigned int*>(&a0);
        unsigned int* a1u = reinterpret_cast<unsigned int*>(&a1);
        unsigned int* bfu = reinterpret_cast<unsigned int*>(&bf);
        const int nrow = (wave * 16 + col) * BS_STRIDE;
        #pragma unroll
        for (int j = 0; j < 8; ++j) {
            const int kpA = 2 * j + ((j >= 4) ? 8 : 0) + (laneHi ? 8 : 0);
            a0u[j] = *(const unsigned int*)&Aw[col * AS_STRIDE + kpA];
            a1u[j] = *(const unsigned int*)&Aw[(col + 16) * AS_STRIDE + kpA];
            bfu[j] = *(const unsigned int*)&Bs[nrow + kpA];
        }
        acc0 = __builtin_amdgcn_wmma_f32_16x16x32_f16(false, a0, false, bf, (short)0, acc0, false, false);
        acc1 = __builtin_amdgcn_wmma_f32_16x16x32_f16(false, a1, false, bf, (short)0, acc1, false, false);
        __syncthreads();
    }
    const float scale = 1.0f / 128.0f;
    float* op = out + ((size_t)b * NTGT + t0) * DZV + wave * 16 + col;
    #pragma unroll
    for (int r = 0; r < 8; ++r) {
        const int m = r + laneHi * 8;
        op[(size_t)m        * DZV] = acc0[r] * scale;
        op[(size_t)(m + 16) * DZV] = acc1[r] * scale;
    }
}

extern "C" void kernel_launch(void* const* d_in, const int* in_sizes, int n_in,
                              void* d_out, int out_size, void* d_ws, size_t ws_size,
                              hipStream_t stream) {
    const float* xg  = (const float*)d_in[0];   // (4,128,128,2)
    const float* zg  = (const float*)d_in[1];   // (4,128,128,128)
    const float* xt  = (const float*)d_in[2];   // (4,2048,2)
    const float* lsp = (const float*)d_in[3];   // (2,)
    float* out = (float*)d_out;                 // (4,2048,128)

    if (ws_size >= ZH_BYTES && d_ws != nullptr) {
        _Float16* zh = (_Float16*)d_ws;
        zpack_kernel<<<dim3(BATCH * NCHUNK), dim3(256), 0, stream>>>(zg, zh);
        setconv_wmma_kernel<<<dim3(BATCH * (NTGT / MT)), dim3(256), 0, stream>>>(
            xg, zh, xt, lsp, out);
    } else {
        setconv_wmma_fallback<<<dim3(BATCH * (NTGT / MT)), dim3(256), 0, stream>>>(
            xg, zg, xt, lsp, out);
    }
}
